// EnhancedBaseModel_80994493268107
// MI455X (gfx1250) — matrix operand order repaired and checked by reading, not compiled, not run on hardware
//
#include <hip/hip_runtime.h>
#include <hip/hip_bf16.h>
#include <math.h>

// ---------------------------------------------------------------------------
// CDNA5 (gfx1250) implementation of EnhancedBaseModel forward pass.
// Convolutions: implicit-GEMM on v_wmma_f32_16x16x32_f16 (f16 in, f32 acc).
// Each wave32 computes a 64(M=OC) x 16(N=positions) output tile:
//  - im2col B-fragment gathered once per K-step (branchless clamped loads,
//    incremental k->(ic,ky,kx) decomposition, full MLP), reused by 4 WMMAs
//  - all 4 weight A-fragments prefetched as aligned float4 (b128) into
//    distinct registers BEFORE any cvt/WMMA -> one deep load clause, and
//    back-to-back WMMAs with no WAR-hazard NOPs between them.
// Bias + LeakyReLU fused in the epilogue.
// ---------------------------------------------------------------------------

typedef __attribute__((ext_vector_type(16))) _Float16 v16h;
typedef __attribute__((ext_vector_type(8)))  float    v8f;

#define MSUB 4   // 16x16 M-subtiles per wave -> 64 output channels per wave

__device__ __forceinline__ int imin(int a, int b) { return a < b ? a : b; }
__device__ __forceinline__ int imax(int a, int b) { return a > b ? a : b; }

// ---------------------------------------------------------------------------
// Implicit-GEMM convolution.
//   GEMM: out[m, n] = sum_k  Wmat[m, k] * Im2col[k, n]
//     m = output channel (OC), n = flattened (b, oy, ox), k = (ic, ky, kx)
//   A fragment (16x32 f16): lane L holds row M = L%16;
//     element e -> K = (L/16)*16 + e                            (same order as B)
//   B fragment (32x16 f16): lane L holds col N = L%16;
//     element e -> K = (L/16)*16 + e                            (same order as A)
//   C/D (16x16 f32): VGPR r -> M = r + 8*(L/16), N = L%16
// Input may be a channel-concat: channels [0,C1) from xA, [C1,IC) from xB.
// ---------------------------------------------------------------------------
__global__ void conv2d_wmma_k(
    const float* __restrict__ xA, const float* __restrict__ xB, int C1,
    const float* __restrict__ w, const float* __restrict__ bias,
    float* __restrict__ out,
    int Nb, int IC, int H, int W,
    int OC, int KH, int KW, int stride, int pad,
    int Ho, int Wo, int relu)
{
    const int lane = threadIdx.x & 31;
    const int waveInBlock = threadIdx.x >> 5;
    const int wavesPerBlock = blockDim.x >> 5;
    const long tile = (long)blockIdx.x * wavesPerBlock + waveInBlock;

    const int MT = 16 * MSUB;
    const int tilesM = (OC + MT - 1) / MT;
    const int Npos   = Nb * Ho * Wo;
    const int tilesN = (Npos + 15) >> 4;
    if (tile >= (long)tilesM * tilesN) return;   // wave-uniform exit (EXEC stays full)

    const int tm = (int)(tile % tilesM);
    const int tn = (int)(tile / tilesM);

    const int h = lane >> 4;     // half-wave index
    const int l = lane & 15;

    // This lane's output position n -> (b, oy, ox)
    const int n = tn * 16 + l;
    int b = 0, oy = 0, ox = 0;
    {
        int nn = (n < Npos) ? n : 0;
        ox = nn % Wo;
        int t = nn / Wo;
        oy = t % Ho;
        b  = t / Ho;
    }

    const int m_base = tm * MT;
    const int KK   = KH * KW;
    const int Ktot = IC * KK;
    const int C2   = IC - C1;

    // Clamped weight-row pointers per M-subtile (rows >= OC read row OC-1;
    // their results are garbage but never stored).
    const float* wrow[MSUB];
#pragma unroll
    for (int ms = 0; ms < MSUB; ++ms) {
        int m = m_base + ms * 16 + l;
        int mc = (m < OC) ? m : (OC - 1);
        wrow[ms] = w + (long)mc * Ktot;
    }

    v8f acc[MSUB];
#pragma unroll
    for (int ms = 0; ms < MSUB; ++ms) acc[ms] = (v8f){};

    const bool kvec = ((Ktot & 3) == 0);   // float4-aligned weight rows

    for (int k0 = 0; k0 < Ktot; k0 += 32) {
        // ---- B fragment: branchless im2col gather (clamped addr + select).
        //      k values per lane are consecutive -> decompose base once and
        //      advance (ic,ky,kx) with a cheap carry chain (no divisions).
        float bv[16];
        bool  bok[16];
        {
            const int kb = k0 + (h << 4);
            int ic = kb / KK;
            int r  = kb - ic * KK;
            int ky = r / KW;
            int kx = r - ky * KW;
#pragma unroll
            for (int e = 0; e < 16; ++e) {
                int k = kb + e;
                bool kv = (k < Ktot) && (n < Npos);
                int iy = oy * stride - pad + ky;
                int ix = ox * stride - pad + kx;
                bool inb = kv && ((unsigned)iy < (unsigned)H) && ((unsigned)ix < (unsigned)W);
                int icc = (ic < IC) ? ic : (IC - 1);          // keep address in-range
                const float* base = (icc < C1)
                    ? (xA + ((long)b * C1 + icc) * H * W)
                    : (xB + ((long)b * C2 + (icc - C1)) * H * W);
                long off = inb ? ((long)iy * W + ix) : 0;     // clamped: always valid
                bv[e]  = base[off];                           // unconditional load
                bok[e] = inb;
                // advance (ic, ky, kx) by one k
                kx += 1;
                bool cx = (kx == KW); kx = cx ? 0 : kx; ky += cx ? 1 : 0;
                bool cy = (ky == KH); ky = cy ? 0 : ky; ic += cy ? 1 : 0;
            }
        }
        v16h bfrag;
#pragma unroll
        for (int e = 0; e < 16; ++e)
            bfrag[e] = bok[e] ? (_Float16)bv[e] : (_Float16)0.0f;

        // ---- A fragments + WMMA ----
        if (kvec && (k0 + 32 <= Ktot)) {
            // Phase 1: prefetch ALL subtiles' weights (16 x b128 in one clause,
            // distinct destination registers -> deep MLP, no WMMA WAR hazards).
            float4 q[MSUB][4];
#pragma unroll
            for (int ms = 0; ms < MSUB; ++ms) {
                const float* wr = wrow[ms] + k0 + (h << 4);
                q[ms][0] = *(const float4*)(wr);
                q[ms][1] = *(const float4*)(wr + 4);
                q[ms][2] = *(const float4*)(wr + 8);
                q[ms][3] = *(const float4*)(wr + 12);
            }
            // Phase 2: convert + back-to-back WMMAs (independent accumulators).
#pragma unroll
            for (int ms = 0; ms < MSUB; ++ms) {
                v16h afrag;
#pragma unroll
                for (int t = 0; t < 4; ++t) {
                    afrag[4 * t + 0] = (_Float16)q[ms][t].x;
                    afrag[4 * t + 1] = (_Float16)q[ms][t].y;
                    afrag[4 * t + 2] = (_Float16)q[ms][t].z;
                    afrag[4 * t + 3] = (_Float16)q[ms][t].w;
                }
                acc[ms] = __builtin_amdgcn_wmma_f32_16x16x32_f16(
                    false, afrag, false, bfrag, (short)0, acc[ms], false, false);
            }
        } else {
            // Tail/odd path (only conv1, Ktot=36): branchless scalar + select.
#pragma unroll
            for (int ms = 0; ms < MSUB; ++ms) {
                v16h afrag;
#pragma unroll
                for (int e = 0; e < 16; ++e) {
                    int k = k0 + (h << 4) + e;
                    bool ok = (k < Ktot);
                    float v = wrow[ms][ok ? k : 0];
                    afrag[e] = ok ? (_Float16)v : (_Float16)0.0f;
                }
                acc[ms] = __builtin_amdgcn_wmma_f32_16x16x32_f16(
                    false, afrag, false, bfrag, (short)0, acc[ms], false, false);
            }
        }
    }

    // ---- Epilogue: bias + LeakyReLU(0.1), scatter to NCHW ----
#pragma unroll
    for (int ms = 0; ms < MSUB; ++ms) {
#pragma unroll
        for (int r = 0; r < 8; ++r) {
            int m = m_base + ms * 16 + r + (h << 3);
            if (m < OC && n < Npos) {
                float v = acc[ms][r] + bias[m];
                if (relu) v = (v >= 0.0f) ? v : 0.1f * v;
                out[(((long)b * OC + m) * Ho + oy) * Wo + ox] = v;
            }
        }
    }
}

// ---------------------------------------------------------------------------
// RoIAlign((H/2, W/2), scale=1, sampling_ratio=2, aligned=False) with
// roi = full image -> separable bilinear sampling + 2x2 mean.
// ---------------------------------------------------------------------------
__global__ void roi_align_half_k(const float* __restrict__ x, float* __restrict__ out,
                                 int Nb, int C, int H, int W)
{
    const int ph = H >> 1, pw = W >> 1;
    const long total = (long)Nb * C * ph * pw;
    long idx = (long)blockIdx.x * blockDim.x + threadIdx.x;
    if (idx >= total) return;

    int j = (int)(idx % pw); long t = idx / pw;
    int i = (int)(t % ph);   t /= ph;
    int c = (int)(t % C);
    int b = (int)(t / C);

    const float binh = (H - 1.0f) / ph;
    const float binw = (W - 1.0f) / pw;
    const float* base = x + ((long)b * C + c) * H * W;

    float acc = 0.0f;
#pragma unroll
    for (int sy = 0; sy < 2; ++sy) {
        float yy = i * binh + (sy + 0.5f) * binh * 0.5f;
        float y0f = floorf(yy);
        float ly = yy - y0f;
        int y0 = imin(imax((int)y0f, 0), H - 1);
        int y1 = imin(y0 + 1, H - 1);
#pragma unroll
        for (int sx = 0; sx < 2; ++sx) {
            float xx = j * binw + (sx + 0.5f) * binw * 0.5f;
            float x0f = floorf(xx);
            float lx = xx - x0f;
            int x0 = imin(imax((int)x0f, 0), W - 1);
            int x1 = imin(x0 + 1, W - 1);
            acc += base[y0 * W + x0] * (1.0f - ly) * (1.0f - lx)
                 + base[y0 * W + x1] * (1.0f - ly) * lx
                 + base[y1 * W + x0] * ly * (1.0f - lx)
                 + base[y1 * W + x1] * ly * lx;
        }
    }
    out[idx] = acc * 0.25f;
}

// out[i] = a[i] + b[i] + c[i]
__global__ void add3_k(const float* __restrict__ a, const float* __restrict__ b,
                       const float* __restrict__ c, float* __restrict__ out, long n)
{
    long i = (long)blockIdx.x * blockDim.x + threadIdx.x;
    if (i < n) out[i] = a[i] + b[i] + c[i];
}

// a[i] += b[i]
__global__ void add2_inplace_k(float* __restrict__ a, const float* __restrict__ b, long n)
{
    long i = (long)blockIdx.x * blockDim.x + threadIdx.x;
    if (i < n) a[i] += b[i];
}

// scores[b,p] = sum_c o1[b,c,p] * Wm[p,c] + bias[p]
__global__ void attn_scores_k(const float* __restrict__ o1, const float* __restrict__ Wm,
                              const float* __restrict__ bv, float* __restrict__ scores,
                              int Nb, int C, int P)
{
    long idx = (long)blockIdx.x * blockDim.x + threadIdx.x;
    if (idx >= (long)Nb * P) return;
    int p = (int)(idx % P);
    int b = (int)(idx / P);
    float s = bv[p];
    for (int c = 0; c < C; ++c)
        s += o1[((long)b * C + c) * P + p] * Wm[(long)p * C + c];
    scores[idx] = s;
}

// row-wise softmax over P (one block per batch row)
__global__ void softmax_rows_k(const float* __restrict__ scores, float* __restrict__ probs, int P)
{
    __shared__ float red[256];
    const int b = blockIdx.x;
    const int tid = threadIdx.x;
    const float* s = scores + (long)b * P;
    float m = -INFINITY;
    for (int p = tid; p < P; p += blockDim.x) m = fmaxf(m, s[p]);
    red[tid] = m; __syncthreads();
    for (int k = 128; k > 0; k >>= 1) {
        if (tid < k) red[tid] = fmaxf(red[tid], red[tid + k]);
        __syncthreads();
    }
    m = red[0]; __syncthreads();
    float sum = 0.0f;
    for (int p = tid; p < P; p += blockDim.x) sum += expf(s[p] - m);
    red[tid] = sum; __syncthreads();
    for (int k = 128; k > 0; k >>= 1) {
        if (tid < k) red[tid] += red[tid + k];
        __syncthreads();
    }
    sum = red[0];
    const float inv = 1.0f / sum;
    for (int p = tid; p < P; p += blockDim.x)
        probs[(long)b * P + p] = expf(s[p] - m) * inv;
}

// prev[b,c,p] *= probs[b,p]   (attention broadcast over channels)
__global__ void mul_attn_k(float* __restrict__ prev, const float* __restrict__ probs,
                           int Nb, int C, int P)
{
    long idx = (long)blockIdx.x * blockDim.x + threadIdx.x;
    if (idx >= (long)Nb * C * P) return;
    int p = (int)(idx % P);
    long t = idx / P;
    int b = (int)(t / C);
    prev[idx] *= probs[(long)b * P + p];
}

// log_softmax over 2 channels: in/out shape [Nb, 2]
__global__ void log_softmax2_k(const float* __restrict__ in, float* __restrict__ out, int Nb)
{
    int b = blockIdx.x * blockDim.x + threadIdx.x;
    if (b >= Nb) return;
    float v0 = in[b * 2 + 0], v1 = in[b * 2 + 1];
    float m = fmaxf(v0, v1);
    float lse = m + logf(expf(v0 - m) + expf(v1 - m));
    out[b * 2 + 0] = v0 - lse;
    out[b * 2 + 1] = v1 - lse;
}

// ---------------------------------------------------------------------------
// Host-side helpers
// ---------------------------------------------------------------------------
static void launch_conv(hipStream_t s, const float* xA, const float* xB, int C1,
                        const float* w, const float* bias, float* out,
                        int Nb, int IC, int H, int W,
                        int OC, int KH, int KW, int stride, int pad, int relu)
{
    int Ho = (H + 2 * pad - KH) / stride + 1;
    int Wo = (W + 2 * pad - KW) / stride + 1;
    int Npos = Nb * Ho * Wo;
    int tilesM = (OC + 16 * MSUB - 1) / (16 * MSUB);
    int tilesN = (Npos + 15) / 16;
    long tiles = (long)tilesM * tilesN;
    const int wavesPerBlock = 4;                   // 128 threads = 4 wave32
    int blocks = (int)((tiles + wavesPerBlock - 1) / wavesPerBlock);
    conv2d_wmma_k<<<blocks, 32 * wavesPerBlock, 0, s>>>(
        xA, xB, C1, w, bias, out, Nb, IC, H, W, OC, KH, KW, stride, pad, Ho, Wo, relu);
}

static void launch_roi(hipStream_t s, const float* x, float* out, int Nb, int C, int H, int W)
{
    long total = (long)Nb * C * (H / 2) * (W / 2);
    int blocks = (int)((total + 255) / 256);
    roi_align_half_k<<<blocks, 256, 0, s>>>(x, out, Nb, C, H, W);
}

static void launch_add3(hipStream_t s, const float* a, const float* b, const float* c,
                        float* out, long n)
{
    add3_k<<<(int)((n + 255) / 256), 256, 0, s>>>(a, b, c, out, n);
}

static void launch_attn(hipStream_t s, const float* o1, const float* aw, const float* ab,
                        float* scores, float* probs, float* prev,
                        int Nb, int Cattn, int Cprev, int P)
{
    long nscore = (long)Nb * P;
    attn_scores_k<<<(int)((nscore + 255) / 256), 256, 0, s>>>(o1, aw, ab, scores, Nb, Cattn, P);
    softmax_rows_k<<<Nb, 256, 0, s>>>(scores, probs, P);
    long nmul = (long)Nb * Cprev * P;
    mul_attn_k<<<(int)((nmul + 255) / 256), 256, 0, s>>>(prev, probs, Nb, Cprev, P);
}

// Workspace layout (float offsets)
static const size_t OFF_A  = 0;          // 12,845,056  (largest activations)
static const size_t OFF_B  = 12845056;   // 12,845,056
static const size_t OFF_C  = 25690112;   // 12,845,056
static const size_t OFF_D  = 38535168;   //  3,211,264  (bo1)
static const size_t OFF_E  = 41746432;   //  3,211,264  (bo2 / o2 carry)
static const size_t OFF_F  = 44957696;   //  3,211,264  (bo3 / o3 carry)
static const size_t OFF_G  = 48168960;   //  1,605,632  (l3)
static const size_t OFF_HX = 49774592;   //    401,408  (aux conv out)
static const size_t OFF_SC = 50176000;   //     50,176  (attention scores)
static const size_t OFF_PR = 50226176;   //     50,176  (attention probs)
static const size_t OFF_H1 = 50276352;   //      4,608  (head p1 out)
static const size_t OFF_H2 = 50280960;   //        128  (head p2 out)
static const size_t OFF_H3 = 50281088;   //          8  (head logits)

extern "C" void kernel_launch(void* const* d_in, const int* in_sizes, int n_in,
                              void* d_out, int out_size, void* d_ws, size_t ws_size,
                              hipStream_t stream)
{
    (void)in_sizes; (void)n_in; (void)out_size; (void)ws_size;
    const float* x = (const float*)d_in[0];
    auto P = [&](int i) { return (const float*)d_in[i]; };

    float* out = (float*)d_out;
    float* ws  = (float*)d_ws;

    float* A  = ws + OFF_A;   float* B  = ws + OFF_B;   float* C  = ws + OFF_C;
    float* D  = ws + OFF_D;   float* E  = ws + OFF_E;   float* F  = ws + OFF_F;
    float* G  = ws + OFF_G;   float* HX = ws + OFF_HX;
    float* SC = ws + OFF_SC;  float* PR = ws + OFF_PR;
    float* H1 = ws + OFF_H1;  float* H2 = ws + OFF_H2;  float* H3 = ws + OFF_H3;

    float* x1 = out + 0;        // 4x64x112x112
    float* x2 = out + 3211264;  // 4x128x56x56
    float* x3 = out + 4816896;  // 4x256x28x28
    float* x4 = out + 5619712;  // 4x512x14x14
    float* x5 = out + 6021120;  // 4x512x7x7
    float* cls = out + 6121472; // 4x2

    const int Nb = 4;

    // ---------------- start_block (b1), H=224 ----------------
    launch_conv(stream, x, nullptr, 4,  P(1), P(2),  A, Nb,   4, 224, 224,  64, 3, 3, 1, 1, 1); // l1
    launch_conv(stream, A, nullptr, 64, P(3), P(4),  B, Nb,  64, 224, 224,  64, 3, 3, 1, 1, 1); // l2
    launch_roi (stream, B, D, Nb, 64, 224, 224);                                                // bo1
    launch_conv(stream, B, nullptr, 64, P(5), P(6),  C, Nb,  64, 224, 224,  64, 1, 1, 1, 0, 1); // aux2 (1x1)
    launch_conv(stream, C, nullptr, 64, P(7), P(8),  E, Nb,  64, 224, 224,  64, 3, 3, 2, 1, 1); // bo2
    add2_inplace_k<<<(int)((12845056L + 255) / 256), 256, 0, stream>>>(A, C, 12845056L);        // l1 += aux2
    launch_conv(stream, A, nullptr, 64, P(9), P(10), F, Nb,  64, 224, 224,  64, 3, 3, 2, 1, 1); // bo3
    launch_add3(stream, D, E, F, x1, 3211264L);                                                 // o1 -> x1

    // ---------------- inter_block1 (b2), H=112, 64->128 ----------------
    launch_conv(stream, x1, nullptr, 64,  P(11), P(12), A, Nb,  64, 112, 112, 128, 3, 3, 1, 1, 1); // l1
    launch_conv(stream, A, nullptr, 128,  P(13), P(14), B, Nb, 128, 112, 112, 128, 3, 3, 1, 1, 1); // l2
    launch_conv(stream, E, F, 64,         P(17), P(18), C, Nb, 128, 112, 112, 128, 3, 3, 1, 1, 1); // prev = conv(cat(o2,o3))
    launch_conv(stream, A, B, 128,        P(15), P(16), E, Nb, 256, 112, 112, 128, 3, 3, 2, 1, 1); // bo2 = conv(cat(l1,l2))
    launch_roi (stream, B, D, Nb, 128, 112, 112);                                                  // bo1
    launch_attn(stream, x1, P(19), P(20), SC, PR, C, Nb, 64, 128, 112 * 112);                      // prev *= attn(o1)
    launch_roi (stream, C, F, Nb, 128, 112, 112);                                                  // bo3
    launch_add3(stream, D, E, F, x2, 1605632L);                                                    // o1 -> x2

    // ---------------- inter_block1 (b3), H=56, 128->256 ----------------
    launch_conv(stream, x2, nullptr, 128, P(21), P(22), A, Nb, 128, 56, 56, 256, 3, 3, 1, 1, 1);   // l1
    launch_conv(stream, A, nullptr, 256,  P(23), P(24), B, Nb, 256, 56, 56, 256, 3, 3, 1, 1, 1);   // l2
    launch_conv(stream, E, F, 128,        P(27), P(28), C, Nb, 256, 56, 56, 256, 3, 3, 1, 1, 1);   // prev
    launch_conv(stream, A, B, 256,        P(25), P(26), E, Nb, 512, 56, 56, 256, 3, 3, 2, 1, 1);   // bo2
    launch_roi (stream, B, D, Nb, 256, 56, 56);                                                    // bo1
    launch_attn(stream, x2, P(29), P(30), SC, PR, C, Nb, 128, 256, 56 * 56);                       // prev *= attn
    launch_roi (stream, C, F, Nb, 256, 56, 56);                                                    // bo3
    launch_add3(stream, D, E, F, x3, 802816L);                                                     // o1 -> x3

    // ---------------- inter_block2 (b4), H=28, 256->512 ----------------
    launch_conv(stream, x3, nullptr, 256, P(31), P(32), A, Nb, 256, 28, 28, 512, 3, 3, 1, 1, 1);   // l1
    launch_conv(stream, A, nullptr, 512,  P(33), P(34), B, Nb, 512, 28, 28, 512, 3, 3, 1, 1, 1);   // l2
    launch_conv(stream, B, nullptr, 512,  P(35), P(36), G, Nb, 512, 28, 28, 512, 3, 3, 1, 1, 1);   // l3
    launch_conv(stream, E, F, 256,        P(41), P(42), C, Nb, 512, 28, 28, 512, 3, 3, 1, 1, 1);   // prev
    launch_conv(stream, A, B, 512,        P(39), P(40), E, Nb, 1024, 28, 28, 512, 3, 3, 2, 1, 1);  // bo2
    launch_roi (stream, B, D, Nb, 512, 28, 28);                                                    // roi(l2)
    launch_conv(stream, G, nullptr, 512,  P(37), P(38), HX, Nb, 512, 28, 28, 512, 3, 3, 2, 1, 1);  // aux(l3)
    add2_inplace_k<<<(int)((401408L + 255) / 256), 256, 0, stream>>>(D, HX, 401408L);              // bo1
    launch_attn(stream, x3, P(43), P(44), SC, PR, C, Nb, 256, 512, 28 * 28);                       // prev *= attn
    launch_roi (stream, C, F, Nb, 512, 28, 28);                                                    // bo3
    launch_add3(stream, D, E, F, x4, 401408L);                                                     // o1 -> x4

    // ---------------- inter_block2 (b5), H=14, 512->512 ----------------
    launch_conv(stream, x4, nullptr, 512, P(45), P(46), A, Nb, 512, 14, 14, 512, 3, 3, 1, 1, 1);   // l1
    launch_conv(stream, A, nullptr, 512,  P(47), P(48), B, Nb, 512, 14, 14, 512, 3, 3, 1, 1, 1);   // l2
    launch_conv(stream, B, nullptr, 512,  P(49), P(50), G, Nb, 512, 14, 14, 512, 3, 3, 1, 1, 1);   // l3
    launch_conv(stream, E, F, 512,        P(55), P(56), C, Nb, 1024, 14, 14, 512, 3, 3, 1, 1, 1);  // prev (IC=1024)
    launch_conv(stream, A, B, 512,        P(53), P(54), E, Nb, 1024, 14, 14, 512, 3, 3, 2, 1, 1);  // bo2
    launch_roi (stream, B, D, Nb, 512, 14, 14);                                                    // roi(l2)
    launch_conv(stream, G, nullptr, 512,  P(51), P(52), HX, Nb, 512, 14, 14, 512, 3, 3, 2, 1, 1);  // aux(l3)
    add2_inplace_k<<<(int)((100352L + 255) / 256), 256, 0, stream>>>(D, HX, 100352L);              // bo1
    launch_attn(stream, x4, P(57), P(58), SC, PR, C, Nb, 512, 512, 14 * 14);                       // prev *= attn
    launch_roi (stream, C, F, Nb, 512, 14, 14);                                                    // bo3
    launch_add3(stream, D, E, F, x5, 100352L);                                                     // o1 -> x5

    // ---------------- head ----------------
    launch_conv(stream, x5, nullptr, 512, P(59), P(60), H1, Nb, 512, 7, 7, 128, 5, 5, 1, 0, 0);    // p1: 7->3
    launch_conv(stream, H1, nullptr, 128, P(61), P(62), H2, Nb, 128, 3, 3,  32, 3, 3, 1, 0, 0);    // p2: 3->1
    launch_conv(stream, H2, nullptr, 32,  P(63), P(64), H3, Nb,  32, 1, 1,   2, 1, 1, 1, 0, 0);    // last 1x1
    log_softmax2_k<<<1, 64, 0, stream>>>(H3, cls, Nb);
}
